// EdgeConvBlock_58162447123322
// MI455X (gfx1250) — matrix operand order repaired and checked by reading, not compiled, not run on hardware
//
#include <hip/hip_runtime.h>
#include <hip/hip_bf16.h>
#include <stdint.h>

// EdgeConv block for MI455X (gfx1250, wave32, WMMA).
// B=4, C_IN=64, C_OUT=128, N=8192, k=16, groups=8.
#define B_   4
#define C_   64      // input channels
#define O_   128     // output channels
#define N_   8192
#define K_   16      // neighbors
#define G_   8       // groupnorm groups (16 ch each == one WMMA N-tile)
#define QPW  8       // queries per wave in the WMMA passes

typedef __attribute__((ext_vector_type(16))) __bf16 v16bf;
typedef __attribute__((ext_vector_type(8)))  float  v8f;
typedef __attribute__((ext_vector_type(2)))  __bf16 bf16x2;

union Frag16 { v16bf v; uint32_t u[8]; };

// ---- bf16 helpers ----
__device__ __forceinline__ uint32_t f32_bf16_rne(float f) {
  uint32_t u = __float_as_uint(f);
  return (u + 0x7fffu + ((u >> 16) & 1u)) >> 16;
}
// packed bf16 subtract: should lower to a single v_pk_add_bf16 (neg modifier)
__device__ __forceinline__ uint32_t bfsub2(uint32_t a, uint32_t b) {
  bf16x2 va = __builtin_bit_cast(bf16x2, a);
  bf16x2 vb = __builtin_bit_cast(bf16x2, b);
  bf16x2 vr = va - vb;
  return __builtin_bit_cast(uint32_t, vr);
}

// ---------------------------------------------------------------------------
// Kernel 1: transpose/convert features to bf16 [B][N][C], convert W to bf16,
// zero the groupnorm stats accumulators (re-zeroed every launch: graph-safe).
// ---------------------------------------------------------------------------
__global__ __launch_bounds__(256) void prep_kernel(
    const float* __restrict__ Fq, const float* __restrict__ Fk,
    const float* __restrict__ W, uint16_t* __restrict__ FqT,
    uint16_t* __restrict__ FkT, uint16_t* __restrict__ Wb,
    float* __restrict__ stats) {
  const int t = blockIdx.x * 256 + threadIdx.x;       // t in [0, B*C*N)
  const int n = t & (N_ - 1);
  const int c = (t >> 13) & (C_ - 1);
  const int b = t >> 19;                              // N_*C_ = 2^19
  const float vq = Fq[(b * C_ + c) * N_ + n];
  const float vk = Fk[(b * C_ + c) * N_ + n];
  const size_t o = (size_t)(b * N_ + n) * C_ + c;
  FqT[o] = (uint16_t)f32_bf16_rne(vq);
  FkT[o] = (uint16_t)f32_bf16_rne(vk);
  if (t < O_ * 2 * C_) Wb[t] = (uint16_t)f32_bf16_rne(W[t]);
  if (t < 2 * B_ * G_) stats[t] = 0.f;
}

// ---------------------------------------------------------------------------
// Kernel 2: brute-force kNN (k=16) with LDS tiling of key points.
// One thread per query, fully predicated register insertion sort.
// ---------------------------------------------------------------------------
__global__ __launch_bounds__(256) void knn_kernel(
    const float* __restrict__ Pq, const float* __restrict__ Pk,
    int* __restrict__ idx) {
  __shared__ float kx[256], ky[256], kz[256];
  const int nb = N_ / 256;
  const int b = blockIdx.x / nb;
  const int n = (blockIdx.x % nb) * 256 + threadIdx.x;
  const float qx = Pq[(b * 3 + 0) * N_ + n];
  const float qy = Pq[(b * 3 + 1) * N_ + n];
  const float qz = Pq[(b * 3 + 2) * N_ + n];

  float bd[K_]; int bi[K_];
#pragma unroll
  for (int t = 0; t < K_; ++t) { bd[t] = 3.0e38f; bi[t] = 0; }

  for (int tile = 0; tile < nb; ++tile) {
    const int j = tile * 256 + threadIdx.x;
    kx[threadIdx.x] = Pk[(b * 3 + 0) * N_ + j];
    ky[threadIdx.x] = Pk[(b * 3 + 1) * N_ + j];
    kz[threadIdx.x] = Pk[(b * 3 + 2) * N_ + j];
    __syncthreads();
    for (int jj = 0; jj < 256; ++jj) {
      const float dx = qx - kx[jj];
      const float dy = qy - ky[jj];
      const float dz = qz - kz[jj];
      const float d = dx * dx + dy * dy + dz * dz;
      if (d < bd[K_ - 1]) {                 // rare: expected ~k*ln(N/k) hits
        const int gj = tile * 256 + jj;
#pragma unroll
        for (int t = K_ - 1; t >= 0; --t) { // predicated shift-insert
          const int   pi    = (t > 0) ? (t - 1) : 0;
          const float dprev = (t > 0) ? bd[pi] : -1.0f;
          const bool  shift = d < dprev;
          const float nd = shift ? dprev : d;
          const int   ni = shift ? bi[pi] : gj;
          if (d < bd[t]) { bd[t] = nd; bi[t] = ni; }
        }
      }
    }
    __syncthreads();
  }
  int* op = idx + ((size_t)b * N_ + n) * K_;
#pragma unroll
  for (int t = 0; t < K_; ++t) op[t] = bi[t];
}

// ---------------------------------------------------------------------------
// Core: one wave computes out[16 neighbors x 128 channels] for one query via
// 8 N-tiles x 4 K-steps of v_wmma_f32_16x16x32_bf16.
// A (16x32 bf16) lane layout: lane holds row (lane&15),
//   elems 0..15 = K 16*(lane>>4)+{0..15} (same order as B).
// B (32x16 bf16): lane holds col (lane&15), elems 0..15 = K 16*(lane>>4)+{0..15}.
// All B offsets are 32B aligned -> direct v16bf LDS loads.
// ---------------------------------------------------------------------------
__device__ __forceinline__ void edge_wmma(
    int b, int n, const uint16_t* __restrict__ FqT,
    const uint16_t* __restrict__ FkT, const int* __restrict__ idx,
    const uint16_t* Wlds, v8f acc[G_]) {
  const int lane = threadIdx.x & 31;
  const int row  = lane & 15;
  const int aoff = (lane >> 4) * 16;
  const int boff = (lane >> 4) * 16;
  const int col  = lane & 15;

  const int j = idx[((size_t)b * N_ + n) * K_ + row];
  const uint16_t* fi = FqT + (size_t)(b * N_ + n) * C_;
  const uint16_t* fj = FkT + (size_t)(b * N_ + j) * C_;

  // Build A fragments: channels [0,64) = nbr - center, [64,128) = center.
  Frag16 A[4];
#pragma unroll
  for (int kb = 0; kb < 4; ++kb) {
    const int cb = kb * 32;
#pragma unroll
    for (int s = 0; s < 2; ++s) {
      const int c0 = cb + aoff + 8 * s;             // 8 contiguous channels
      if (cb < C_) {                                 // diff region
        const uint4 vj = *(const uint4*)(fj + c0);
        const uint4 vi = *(const uint4*)(fi + c0);
        A[kb].u[4 * s + 0] = bfsub2(vj.x, vi.x);
        A[kb].u[4 * s + 1] = bfsub2(vj.y, vi.y);
        A[kb].u[4 * s + 2] = bfsub2(vj.z, vi.z);
        A[kb].u[4 * s + 3] = bfsub2(vj.w, vi.w);
      } else {                                       // broadcast-center region
        const uint4 vi = *(const uint4*)(fi + (c0 - C_));
        A[kb].u[4 * s + 0] = vi.x;
        A[kb].u[4 * s + 1] = vi.y;
        A[kb].u[4 * s + 2] = vi.z;
        A[kb].u[4 * s + 3] = vi.w;
      }
    }
  }

#pragma unroll
  for (int nt = 0; nt < G_; ++nt) {
    v8f c = {};
#pragma unroll
    for (int kb = 0; kb < 4; ++kb) {
      // 32B-aligned: (nt*16+col)*256B + kb*64B + {0,32}B
      const v16bf* wp =
          (const v16bf*)(Wlds + (nt * 16 + col) * (2 * C_) + kb * 32 + boff);
      c = __builtin_amdgcn_wmma_f32_16x16x32_bf16(false, A[kb].v, false, *wp,
                                                  (short)0, c, false, false);
    }
    acc[nt] = c;
  }
}

// ---------------------------------------------------------------------------
// Kernel 3 (pass 1): WMMA + per-group sum / sumsq accumulation for GroupNorm.
// ---------------------------------------------------------------------------
__global__ __launch_bounds__(128) void pass1_kernel(
    const uint16_t* __restrict__ FqT, const uint16_t* __restrict__ FkT,
    const uint16_t* __restrict__ Wb, const int* __restrict__ idx,
    float* __restrict__ stats) {
  __shared__ __align__(32) uint16_t Wlds[O_ * 2 * C_];
  {
    const uint4* src = (const uint4*)Wb;
    uint4* dst = (uint4*)Wlds;
    for (int i = threadIdx.x; i < (O_ * 2 * C_) / 8; i += blockDim.x)
      dst[i] = src[i];
  }
  __syncthreads();

  const int wave = blockIdx.x * (blockDim.x >> 5) + (threadIdx.x >> 5);
  const int q0 = wave * QPW;
  const int b = q0 >> 13;                     // N_ = 2^13, same b for QPW queries
  const int lane = threadIdx.x & 31;

  float sum[G_], ssq[G_];
#pragma unroll
  for (int g = 0; g < G_; ++g) { sum[g] = 0.f; ssq[g] = 0.f; }

  for (int q = q0; q < q0 + QPW; ++q) {
    const int n = q & (N_ - 1);
    v8f acc[G_];
    edge_wmma(b, n, FqT, FkT, idx, Wlds, acc);
#pragma unroll
    for (int g = 0; g < G_; ++g)
#pragma unroll
      for (int r = 0; r < 8; ++r) {
        const float x = acc[g][r];
        sum[g] += x; ssq[g] += x * x;
      }
  }
#pragma unroll
  for (int g = 0; g < G_; ++g) {
    float s = sum[g], z = ssq[g];
#pragma unroll
    for (int d = 16; d >= 1; d >>= 1) {
      s += __shfl_xor(s, d, 32);
      z += __shfl_xor(z, d, 32);
    }
    if (lane == 0) {
      atomicAdd(&stats[(b * G_ + g) * 2 + 0], s);
      atomicAdd(&stats[(b * G_ + g) * 2 + 1], z);
    }
  }
}

// ---------------------------------------------------------------------------
// Kernel 4: turn (sum, sumsq) into (mean, invstd).
// ---------------------------------------------------------------------------
__global__ void gn_finalize_kernel(const float* __restrict__ stats,
                                   float* __restrict__ gnp) {
  const int t = threadIdx.x;
  if (t < B_ * G_) {
    const float cnt = (float)(O_ / G_) * (float)N_ * (float)K_;
    const float mean = stats[t * 2 + 0] / cnt;
    const float var  = stats[t * 2 + 1] / cnt - mean * mean;
    gnp[t * 2 + 0] = mean;
    gnp[t * 2 + 1] = rsqrtf(var + 1e-5f);
  }
}

// ---------------------------------------------------------------------------
// Kernel 5 (pass 2): recompute WMMA, fuse affine-GN -> max over k -> ReLU.
// C/D layout: lane holds channel (lane&15), rows r + 8*(lane>>4); combine the
// two row-halves with shfl_xor(16).
// ---------------------------------------------------------------------------
__global__ __launch_bounds__(128) void pass2_kernel(
    const uint16_t* __restrict__ FqT, const uint16_t* __restrict__ FkT,
    const uint16_t* __restrict__ Wb, const int* __restrict__ idx,
    const float* __restrict__ gnp, const float* __restrict__ gamma,
    const float* __restrict__ beta, float* __restrict__ out) {
  __shared__ __align__(32) uint16_t Wlds[O_ * 2 * C_];
  {
    const uint4* src = (const uint4*)Wb;
    uint4* dst = (uint4*)Wlds;
    for (int i = threadIdx.x; i < (O_ * 2 * C_) / 8; i += blockDim.x)
      dst[i] = src[i];
  }
  __syncthreads();

  const int wave = blockIdx.x * (blockDim.x >> 5) + (threadIdx.x >> 5);
  const int q0 = wave * QPW;
  const int b = q0 >> 13;
  const int lane = threadIdx.x & 31;
  const int col = lane & 15;

  // per-channel affine hoisted out of the query loop (b fixed per wave)
  float scale[G_], bias[G_];
#pragma unroll
  for (int g = 0; g < G_; ++g) {
    const int ch = g * 16 + col;
    const float mean = gnp[(b * G_ + g) * 2 + 0];
    const float inv  = gnp[(b * G_ + g) * 2 + 1];
    const float sc = gamma[ch] * inv;
    scale[g] = sc;
    bias[g] = beta[ch] - mean * sc;
  }

  for (int q = q0; q < q0 + QPW; ++q) {
    const int n = q & (N_ - 1);
    v8f acc[G_];
    edge_wmma(b, n, FqT, FkT, idx, Wlds, acc);
#pragma unroll
    for (int g = 0; g < G_; ++g) {
      float m = -3.0e38f;
#pragma unroll
      for (int r = 0; r < 8; ++r)
        m = fmaxf(m, acc[g][r] * scale[g] + bias[g]);
      m = fmaxf(m, __shfl_xor(m, 16, 32));      // merge the two row halves
      if (lane < 16)
        out[((size_t)b * O_ + g * 16 + col) * N_ + n] = fmaxf(m, 0.f);
    }
  }
}

// ---------------------------------------------------------------------------
// Workspace layout (bytes), total ~10.6 MB:
//   [0, 4M)        FqT  bf16 [B][N][C]
//   [4M, 8M)       FkT  bf16 [B][N][C]
//   [8M, +32K)     Wb   bf16 [O][2C]
//   [+32K, +2M)    idx  int  [B][N][K]
//   then stats (64 f32) and gn params (64 f32)
// ---------------------------------------------------------------------------
extern "C" void kernel_launch(void* const* d_in, const int* in_sizes, int n_in,
                              void* d_out, int out_size, void* d_ws, size_t ws_size,
                              hipStream_t stream) {
  (void)in_sizes; (void)n_in; (void)out_size; (void)ws_size;
  const float* Fq    = (const float*)d_in[0];
  const float* Fk    = (const float*)d_in[1];
  const float* Pq    = (const float*)d_in[2];
  const float* Pk    = (const float*)d_in[3];
  const float* W     = (const float*)d_in[4];
  const float* gamma = (const float*)d_in[5];
  const float* beta  = (const float*)d_in[6];
  // d_in[7] is k; fixed to 16 here.

  char* ws = (char*)d_ws;
  uint16_t* FqT  = (uint16_t*)(ws);
  uint16_t* FkT  = (uint16_t*)(ws + 4194304);
  uint16_t* Wb   = (uint16_t*)(ws + 8388608);
  int*      idx  = (int*)    (ws + 8421376);
  float*    stats= (float*)  (ws + 10518528);
  float*    gnp  = (float*)  (ws + 10518784);
  float*    out  = (float*)d_out;

  prep_kernel<<<(B_ * N_ * C_) / 256, 256, 0, stream>>>(Fq, Fk, W, FqT, FkT, Wb, stats);
  knn_kernel<<<B_ * (N_ / 256), 256, 0, stream>>>(Pq, Pk, idx);
  const int blocks = (B_ * N_) / (QPW * 4);   // 4 waves per 128-thread block
  pass1_kernel<<<blocks, 128, 0, stream>>>(FqT, FkT, Wb, idx, stats);
  gn_finalize_kernel<<<1, 64, 0, stream>>>(stats, gnp);
  pass2_kernel<<<blocks, 128, 0, stream>>>(FqT, FkT, Wb, idx, gnp, gamma, beta, out);
}
